// Attention_15204184228001
// MI455X (gfx1250) — matrix operand order repaired and checked by reading, not compiled, not run on hardware
//
#include <hip/hip_runtime.h>
#include <math.h>

// ---------------------------------------------------------------------------
// MI455X (gfx1250) implementation: all GEMMs via v_wmma_f32_16x16x32_bf16,
// fp32 accumulate, wave32 tile layouts per CDNA5 ISA 7.12.2.
// ---------------------------------------------------------------------------

typedef __bf16 bf16_t;
typedef __attribute__((ext_vector_type(16))) __bf16 v16bf;
typedef __attribute__((ext_vector_type(8)))  __bf16 v8bf;
typedef __attribute__((ext_vector_type(8)))  float  v8f;

#define DIMC  512
#define NSEQ  256
#define NBAT  64
#define NHEAD 8
#define DH    64

__device__ __forceinline__ v8f wmma_bf16(v16bf a, v16bf b, v8f c) {
  // D = A(16x32 bf16) * B(32x16 bf16) + C(16x16 f32)
  return __builtin_amdgcn_wmma_f32_16x16x32_bf16(
      /*neg_a=*/false, a, /*neg_b=*/false, b,
      /*c_mod=*/(short)0, c, /*reuse_a=*/false, /*reuse_b=*/false);
}

// A-fragment: lane (row = lane&15) holds K runs [kb + hs*16, +8) and [kb+hs*16+8, +8)
__device__ __forceinline__ v16bf make_a(const bf16_t* run1, const bf16_t* run2) {
  v8bf lo = *(const v8bf*)run1;
  v8bf hi = *(const v8bf*)run2;
  v16bf a;
#pragma unroll
  for (int i = 0; i < 8; ++i) { a[i] = lo[i]; a[i + 8] = hi[i]; }
  return a;
}

// ---------------------------------------------------------------------------
// K0: convert + transpose weights to bf16 so B-fragments are contiguous-K.
// WinT[j][k] = Win[k][j], WoutT[j][k] = Wout[k][j]
// ---------------------------------------------------------------------------
__global__ __launch_bounds__(256)
void k_prep_weights(const float* __restrict__ Win, const float* __restrict__ Wout,
                    bf16_t* __restrict__ WinT, bf16_t* __restrict__ WoutT) {
  int idx = blockIdx.x * 256 + threadIdx.x;   // 0 .. 512*512-1
  int j = idx >> 9;
  int kk = idx & 511;
  WinT [idx] = (bf16_t)Win [kk * DIMC + j];
  WoutT[idx] = (bf16_t)Wout[kk * DIMC + j];
}

// ---------------------------------------------------------------------------
// K1: f_x = LayerNorm(x) @ W_in, for x in {q,k,v} (blockIdx.y).
// One block = 16 rows (b*256+n flattened). 8 waves cover 32 col-tiles of 16.
// Outputs: fq/fk bf16 [h][b][n][64] (row-major K), fvT bf16 [h][b][d][n].
// ---------------------------------------------------------------------------
__global__ __launch_bounds__(256)
void k_transform(const float* __restrict__ q, const float* __restrict__ k,
                 const float* __restrict__ v, const float* __restrict__ g,
                 const float* __restrict__ beta, const bf16_t* __restrict__ WinT,
                 bf16_t* __restrict__ fq, bf16_t* __restrict__ fk,
                 bf16_t* __restrict__ fvT) {
  const int which = blockIdx.y;
  const float* src = (which == 0) ? q : (which == 1) ? k : v;
  const int rowTile = blockIdx.x;            // 0..1023 (16384 rows / 16)
  const int tid = threadIdx.x;
  const int lane = tid & 31, wave = tid >> 5;
  const int hs = lane >> 4, l15 = lane & 15;

  __shared__ float xf[16][DIMC + 4];
  __shared__ __align__(16) bf16_t xn[16][DIMC + 16];
  __shared__ float red1[16][16], red2[16][16];
  __shared__ float mu_s[16], rs_s[16];

  const float* srcBase = src + (size_t)rowTile * 16 * DIMC;
  for (int i = tid; i < 16 * DIMC; i += 256) {
    int r = i >> 9, c = i & (DIMC - 1);
    xf[r][c] = srcBase[i];
  }
  __syncthreads();
  {
    int r = tid >> 4, s16 = tid & 15;
    float s1 = 0.f, s2 = 0.f;
    for (int c = s16; c < DIMC; c += 16) { float x = xf[r][c]; s1 += x; s2 += x * x; }
    red1[r][s16] = s1; red2[r][s16] = s2;
  }
  __syncthreads();
  if (tid < 16) {
    float s1 = 0.f, s2 = 0.f;
#pragma unroll
    for (int i = 0; i < 16; ++i) { s1 += red1[tid][i]; s2 += red2[tid][i]; }
    float mu  = s1 * (1.0f / DIMC);
    float var = s2 * (1.0f / DIMC) - mu * mu;
    mu_s[tid] = mu;
    rs_s[tid] = rsqrtf(var + 1e-5f);
  }
  __syncthreads();
  for (int i = tid; i < 16 * DIMC; i += 256) {
    int r = i >> 9, c = i & (DIMC - 1);
    float y = (xf[r][c] - mu_s[r]) * rs_s[r] * g[c] + beta[c];
    xn[r][c] = (bf16_t)y;
  }
  __syncthreads();

  for (int ct = wave; ct < 32; ct += 8) {
    v8f acc = {};
    const bf16_t* bcol = WinT + (size_t)(ct * 16 + l15) * DIMC + hs * 16;
#pragma unroll
    for (int kt = 0; kt < 16; ++kt) {
      int kb = kt * 32;
      v16bf A  = make_a(&xn[l15][kb + hs * 16], &xn[l15][kb + hs * 16 + 8]);
      v16bf Bm = *(const v16bf*)(bcol + kb);
      acc = wmma_bf16(A, Bm, acc);
    }
    int colG = ct * 16 + l15;            // 0..511 inner
    int h = colG >> 6, d = colG & 63;
#pragma unroll
    for (int j = 0; j < 8; ++j) {
      int m = j + hs * 8;                // row within tile (C layout: M = j + 8*(lane/16))
      int gRow = rowTile * 16 + m;       // b*256 + n
      int b = gRow >> 8, n = gRow & 255;
      bf16_t val = (bf16_t)acc[j];
      size_t hb = (size_t)(h * NBAT + b);
      if (which == 0)      fq [(hb * NSEQ + n) * DH + d]  = val;
      else if (which == 1) fk [(hb * NSEQ + n) * DH + d]  = val;
      else                 fvT[(hb * DH + d) * NSEQ + n]  = val;
    }
  }
}

// ---------------------------------------------------------------------------
// K2: per-sample chunk statistics. One block per sample s in [0,1024):
// s<512 -> f_q sample, else f_k sample. Produces Vs[s], Cs[s].
// Covariance 64x64 = centered^T @ centered / 255 via WMMA SYRK from LDS.
// ---------------------------------------------------------------------------
__global__ __launch_bounds__(256)
void k_chunkstats(const bf16_t* __restrict__ fq, const bf16_t* __restrict__ fk,
                  float* __restrict__ Vs, float* __restrict__ Cs) {
  const int s = blockIdx.x;
  const bf16_t* E = ((s < 512) ? fq : fk) + (size_t)(s & 511) * NSEQ * DH;
  const int tid = threadIdx.x;
  const int lane = tid & 31, wave = tid >> 5;
  const int hs = lane >> 4, l15 = lane & 15;

  __shared__ __align__(16) bf16_t cT[DH][NSEQ + 16];  // centered, transposed [d][n]
  __shared__ float p1[DH][4], p2[DH][4];
  __shared__ float mu_s[DH], relu_s[DH];
  __shared__ float tot_s, diag_s, V_s;

  const int d = tid & 63, quarter = tid >> 6;
  float s1 = 0.f, s2 = 0.f;
  for (int n = quarter * 64; n < quarter * 64 + 64; ++n) {
    float x = (float)E[n * DH + d];
    s1 += x; s2 += x * x;
  }
  p1[d][quarter] = s1; p2[d][quarter] = s2;
  if (tid == 0) { tot_s = 0.f; diag_s = 0.f; }
  __syncthreads();
  if (tid < DH) {
    float S1 = p1[tid][0] + p1[tid][1] + p1[tid][2] + p1[tid][3];
    float S2 = p2[tid][0] + p2[tid][1] + p2[tid][2] + p2[tid][3];
    float mu  = S1 * (1.0f / NSEQ);
    float var = (S2 - (float)NSEQ * mu * mu) * (1.0f / (NSEQ - 1));  // ddof=1
    mu_s[tid]   = mu;
    relu_s[tid] = fmaxf(0.f, 1.0f - sqrtf(var + 1e-8f));
  }
  __syncthreads();
  for (int n = quarter * 64; n < quarter * 64 + 64; ++n) {
    cT[d][n] = (bf16_t)((float)E[n * DH + d] - mu_s[d]);
  }
  if (tid == 0) {
    float acc = 0.f;
#pragma unroll
    for (int i = 0; i < DH; ++i) acc += relu_s[i];
    V_s = acc * (1.0f / DH);
  }
  __syncthreads();

  float tot = 0.f, diag = 0.f;
  for (int t = wave; t < 16; t += 8) {          // 4x4 tiles of 16x16 over features
    int ti = t >> 2, tj = t & 3;
    v8f acc = {};
#pragma unroll
    for (int kt = 0; kt < 8; ++kt) {            // K = 256 over n
      int kb = kt * 32;
      v16bf A  = make_a(&cT[ti * 16 + l15][kb + hs * 16],
                        &cT[ti * 16 + l15][kb + hs * 16 + 8]);
      v16bf Bm = *(const v16bf*)(&cT[tj * 16 + l15][kb + hs * 16]);
      acc = wmma_bf16(A, Bm, acc);
    }
#pragma unroll
    for (int j = 0; j < 8; ++j) {
      int ig = ti * 16 + j + hs * 8;
      int jg = tj * 16 + l15;
      float c  = acc[j] * (1.0f / (NSEQ - 1));
      float sq = c * c;
      tot += sq;
      if (ig == jg) diag += sq;
    }
  }
  atomicAdd(&tot_s, tot);
  atomicAdd(&diag_s, diag);
  __syncthreads();
  if (tid == 0) {
    Vs[s] = V_s;
    Cs[s] = (tot_s - diag_s) * (1.0f / DH);
  }
}

// ---------------------------------------------------------------------------
// K3: fused attention per (h,b): dots = cos_w * (fq fk^T) * rqn*rkn + mix,
// staged in LDS as bf16, then out = dots @ fv via fvT, written to attn
// bf16 [b][n][h*64+d] (row-major K for final projection).
// Dynamic LDS: 256 x 264 bf16 = 135168 B (within 320KB WGP LDS).
// ---------------------------------------------------------------------------
#define DOTS_STRIDE (NSEQ + 8)

__global__ __launch_bounds__(256)
void k_attn(const bf16_t* __restrict__ fq, const bf16_t* __restrict__ fk,
            const bf16_t* __restrict__ fvT, const float* __restrict__ Vs,
            const float* __restrict__ Cs, const float* __restrict__ cov_logit,
            const float* __restrict__ var_logit, bf16_t* __restrict__ attn) {
  const int hb = blockIdx.x;                 // h*64 + b
  const int tid = threadIdx.x;
  const int lane = tid & 31, wave = tid >> 5;
  const int hs = lane >> 4, l15 = lane & 15;

  extern __shared__ __align__(16) char smem_dyn[];
  bf16_t (*dotsS)[DOTS_STRIDE] = (bf16_t(*)[DOTS_STRIDE])smem_dyn;

  __shared__ float rqn_s[NSEQ], rkn_s[NSEQ];  // reciprocal norms
  __shared__ float cosw_s, mix_s;

  const bf16_t* fqB = fq  + (size_t)hb * NSEQ * DH;
  const bf16_t* fkB = fk  + (size_t)hb * NSEQ * DH;
  const bf16_t* fvB = fvT + (size_t)hb * DH * NSEQ;

  { // reciprocal norms, one row per thread; vectorized b128 reads
    float sq = 0.f, sk = 0.f;
    const bf16_t* pq = fqB + tid * DH;
    const bf16_t* pk = fkB + tid * DH;
#pragma unroll
    for (int c8 = 0; c8 < 8; ++c8) {
      v8bf aq = *(const v8bf*)(pq + c8 * 8);
      v8bf ak = *(const v8bf*)(pk + c8 * 8);
#pragma unroll
      for (int i = 0; i < 8; ++i) {
        float a = (float)aq[i]; sq += a * a;
        float b = (float)ak[i]; sk += b * b;
      }
    }
    rqn_s[tid] = rsqrtf(sq);   // 1/|q_n|
    rkn_s[tid] = rsqrtf(sk);   // 1/|k_m|
  }
  if (tid == 0) {
    float cw = 1.f / (1.f + __expf(-cov_logit[0]));
    float vw = 1.f / (1.f + __expf(-var_logit[0]));
    int c = hb >> 3;                         // chunk over flattened h*B, CHUNK=8
    float vq = 0.f, vk = 0.f, cq = 0.f, ck = 0.f;
#pragma unroll
    for (int i = 0; i < 8; ++i) {
      vq += Vs[c * 8 + i];        vk += Vs[512 + c * 8 + i];
      cq += Cs[c * 8 + i];        ck += Cs[512 + c * 8 + i];
    }
    vq *= 0.125f; vk *= 0.125f; cq *= 0.125f; ck *= 0.125f;
    cosw_s = 1.f - cw - vw;
    mix_s  = cw * (cq * ck) + vw * (vq * vk);
  }
  __syncthreads();
  const float cosw = cosw_s, mix = mix_s;

  // GEMM1: 256x256 dots, K = 64. Each wave: 2 tn-tiles, A hoisted across 16 tm.
#pragma unroll
  for (int ts = 0; ts < 2; ++ts) {
    const int tn = wave * 2 + ts;
    const bf16_t* arow = fqB + (tn * 16 + l15) * DH;
    v16bf A0 = make_a(arow +      hs * 16, arow +  8 + hs * 16);
    v16bf A1 = make_a(arow + 32 + hs * 16, arow + 40 + hs * 16);
    for (int tm = 0; tm < 16; ++tm) {
      const bf16_t* brow = fkB + (tm * 16 + l15) * DH + hs * 16;
      v16bf B0 = *(const v16bf*)(brow);
      v16bf B1 = *(const v16bf*)(brow + 32);
      v8f acc = {};
      acc = wmma_bf16(A0, B0, acc);
      acc = wmma_bf16(A1, B1, acc);
      int mg = tm * 16 + l15;
      float rkm = cosw * rkn_s[mg];
#pragma unroll
      for (int j = 0; j < 8; ++j) {
        int ng = tn * 16 + j + hs * 8;
        float val = acc[j] * rqn_s[ng] * rkm + mix;   // pure FMA epilogue
        dotsS[ng][mg] = (bf16_t)val;
      }
    }
  }
  __syncthreads();

  // GEMM2: 256x64 out = dots @ fv, K = 256. Each wave: 2 tn-tiles,
  // 8 A-fragments (LDS) hoisted across 4 td-tiles.
  const int h = hb >> 6, b = hb & 63;
#pragma unroll
  for (int ts = 0; ts < 2; ++ts) {
    const int tn = wave * 2 + ts;
    const bf16_t* arow = &dotsS[tn * 16 + l15][0];
    v16bf Afr[8];
#pragma unroll
    for (int kt = 0; kt < 8; ++kt) {
      int kb = kt * 32;
      Afr[kt] = make_a(arow + kb + hs * 16, arow + kb + hs * 16 + 8);
    }
    for (int td = 0; td < 4; ++td) {
      const bf16_t* brow = fvB + (td * 16 + l15) * NSEQ + hs * 16;
      v8f acc = {};
#pragma unroll
      for (int kt = 0; kt < 8; ++kt) {
        v16bf Bm = *(const v16bf*)(brow + kt * 32);
        acc = wmma_bf16(Afr[kt], Bm, acc);
      }
      int col = h * DH + td * 16 + l15;        // inner index in [0,512)
#pragma unroll
      for (int j = 0; j < 8; ++j) {
        int ng = tn * 16 + j + hs * 8;
        attn[((size_t)(b * NSEQ + ng)) * DIMC + col] = (bf16_t)acc[j];
      }
    }
  }
}

// ---------------------------------------------------------------------------
// K4: out = attn @ W_out + b_out (fp32 output). Same tiling as K1's GEMM.
// ---------------------------------------------------------------------------
__global__ __launch_bounds__(256)
void k_outproj(const bf16_t* __restrict__ attn, const bf16_t* __restrict__ WoutT,
               const float* __restrict__ b_out, float* __restrict__ out) {
  const int rt = blockIdx.x;                 // 0..1023
  const int tid = threadIdx.x;
  const int lane = tid & 31, wave = tid >> 5;
  const int hs = lane >> 4, l15 = lane & 15;

  for (int ct = wave; ct < 32; ct += 8) {
    v8f acc = {};
    const bf16_t* arow = attn  + (size_t)(rt * 16 + l15) * DIMC;
    const bf16_t* bcol = WoutT + (size_t)(ct * 16 + l15) * DIMC + hs * 16;
#pragma unroll
    for (int kt = 0; kt < 16; ++kt) {
      int kb = kt * 32;
      v16bf A  = make_a(arow + kb + hs * 16, arow + kb + hs * 16 + 8);
      v16bf Bm = *(const v16bf*)(bcol + kb);
      acc = wmma_bf16(A, Bm, acc);
    }
    int col = ct * 16 + l15;
    float bias = b_out[col];
#pragma unroll
    for (int j = 0; j < 8; ++j) {
      int row = rt * 16 + j + hs * 8;
      out[(size_t)row * DIMC + col] = acc[j] + bias;
    }
  }
}

// ---------------------------------------------------------------------------
extern "C" void kernel_launch(void* const* d_in, const int* in_sizes, int n_in,
                              void* d_out, int out_size, void* d_ws, size_t ws_size,
                              hipStream_t stream) {
  const float* q         = (const float*)d_in[0];
  const float* k         = (const float*)d_in[1];
  const float* v         = (const float*)d_in[2];
  const float* ln_gamma  = (const float*)d_in[3];
  const float* ln_beta   = (const float*)d_in[4];
  const float* W_in      = (const float*)d_in[5];
  const float* W_out     = (const float*)d_in[6];
  const float* b_out     = (const float*)d_in[7];
  const float* cov_logit = (const float*)d_in[8];
  const float* var_logit = (const float*)d_in[9];
  float* out = (float*)d_out;

  char* ws = (char*)d_ws;
  size_t off = 0;
  auto walloc = [&](size_t bytes) {
    char* p = ws + off;
    off += (bytes + 255) & ~(size_t)255;
    return p;
  };
  bf16_t* WinT  = (bf16_t*)walloc((size_t)DIMC * DIMC * 2);
  bf16_t* WoutT = (bf16_t*)walloc((size_t)DIMC * DIMC * 2);
  bf16_t* fq    = (bf16_t*)walloc((size_t)NHEAD * NBAT * NSEQ * DH * 2);
  bf16_t* fk    = (bf16_t*)walloc((size_t)NHEAD * NBAT * NSEQ * DH * 2);
  bf16_t* fvT   = (bf16_t*)walloc((size_t)NHEAD * NBAT * DH * NSEQ * 2);
  bf16_t* attn  = (bf16_t*)walloc((size_t)NBAT * NSEQ * DIMC * 2);
  float*  Vs    = (float*)walloc(1024 * 4);
  float*  Cs    = (float*)walloc(1024 * 4);

  const size_t dotsBytes = (size_t)NSEQ * DOTS_STRIDE * sizeof(bf16_t);  // 135168 B
  // Raise the dynamic-LDS cap (ISA allows 320 KB per workgroup on gfx1250).
  (void)hipFuncSetAttribute((const void*)k_attn,
                            hipFuncAttributeMaxDynamicSharedMemorySize,
                            (int)dotsBytes);

  k_prep_weights<<<dim3(1024), 256, 0, stream>>>(W_in, W_out, WinT, WoutT);
  k_transform<<<dim3(1024, 3), 256, 0, stream>>>(q, k, v, ln_gamma, ln_beta,
                                                 WinT, fq, fk, fvT);
  k_chunkstats<<<dim3(1024), 256, 0, stream>>>(fq, fk, Vs, Cs);
  k_attn<<<dim3(NHEAD * NBAT), 256, dotsBytes, stream>>>(fq, fk, fvT, Vs, Cs,
                                                         cov_logit, var_logit, attn);
  k_outproj<<<dim3(1024), 256, 0, stream>>>(attn, WoutT, b_out, out);
}
